// SelfAttention_14053132993159
// MI455X (gfx1250) — matrix operand order repaired and checked by reading, not compiled, not run on hardware
//
#include <hip/hip_runtime.h>

typedef __attribute__((ext_vector_type(16))) _Float16 v16h;
typedef __attribute__((ext_vector_type(8)))  _Float16 v8h;
typedef __attribute__((ext_vector_type(8)))  float    v8f;
typedef __attribute__((ext_vector_type(4)))  unsigned int u32x4;
typedef __attribute__((ext_vector_type(8)))  int      i32x8;
typedef __attribute__((ext_vector_type(4)))  int      i32x4;

#define NB     8
#define NC     64
#define NCQ    8
#define NPIX   4096
#define NTIL   (NPIX / 16)         // 256 16-row tiles per batch
#define VCHUNK 128                 // j columns staged per TDM transfer
#define NCHUNK (NPIX / VCHUNK)     // 32 chunks
#define KCH    (VCHUNK / 16)       // 8 pre-packed K tiles per chunk
// pre-packed operand image: [tile][lane(32)][16 halves] = 1KB per 16 rows
#define OPTILE (32 * 16)           // halves per packed tile

// ---------------------------------------------------------------------------
// Kernel 1: per-pixel 1x1-conv projections. One thread per (b, n).
// qA/kA are stored as exact WMMA operand register images for a 16-row tile:
// lane slot (n%16) gets the 8 channel values in elements 0..7, zeros
// elsewhere; lane slots 16..31 are all zero (d=8 padded to K=32).
// v stored [b][c][n] f16 (TDM-friendly 2D tensor, contiguous in j).
// ---------------------------------------------------------------------------
__global__ __launch_bounds__(256) void qkv_proj_kernel(
    const float* __restrict__ x,
    const float* __restrict__ wq, const float* __restrict__ bq,
    const float* __restrict__ wk, const float* __restrict__ bk,
    const float* __restrict__ wv, const float* __restrict__ bv,
    _Float16* __restrict__ qA, _Float16* __restrict__ kA,
    _Float16* __restrict__ vm)
{
    const int idx = blockIdx.x * blockDim.x + threadIdx.x;   // b*NPIX + n
    const int b = idx >> 12;
    const int n = idx & (NPIX - 1);

    float xc[NC];
    const float* xb = x + ((size_t)b * NC) * NPIX + n;
#pragma unroll
    for (int c = 0; c < NC; ++c) xc[c] = xb[(size_t)c * NPIX];

    v16h qa, ka, za;
#pragma unroll
    for (int e = 0; e < 16; ++e) { qa[e] = (_Float16)0.f; ka[e] = (_Float16)0.f; za[e] = (_Float16)0.f; }
#pragma unroll
    for (int d = 0; d < NCQ; ++d) {
        float aq = bq[d];
        float ak = bk[d];
#pragma unroll
        for (int c = 0; c < NC; ++c) {
            aq = fmaf(wq[d * NC + c], xc[c], aq);
            ak = fmaf(wk[d * NC + c], xc[c], ak);
        }
        qa[d] = (_Float16)aq;
        ka[d] = (_Float16)ak;
    }
    const size_t slot = ((size_t)(b * NTIL + (n >> 4)) * 32 + (n & 15)) * 16;
    *(v16h*)(qA + slot)       = qa;
    *(v16h*)(qA + slot + 256) = za;    // lane slot (n%16)+16: zeros
    *(v16h*)(kA + slot)       = ka;
    *(v16h*)(kA + slot + 256) = za;

    for (int o = 0; o < NC; ++o) {
        float av = bv[o];
#pragma unroll
        for (int c = 0; c < NC; ++c) av = fmaf(wv[o * NC + c], xc[c], av);
        vm[((size_t)b * NC + o) * NPIX + n] = (_Float16)av;
    }
}

// ---------------------------------------------------------------------------
// TDM 2D descriptor (6-arg builtin on this toolchain). data_size = 2 bytes.
// ---------------------------------------------------------------------------
__device__ __forceinline__ void tdm_load_2d(const _Float16* gtile,
                                            unsigned int lds_addr,
                                            int tile0, int tile1,
                                            int tdim0, int tdim1, int stride0)
{
    const unsigned long long ga = (unsigned long long)(uintptr_t)gtile;
    u32x4 g0;
    g0[0] = 1u;                                            // count=1, no gather
    g0[1] = lds_addr;                                      // LDS byte offset
    g0[2] = (unsigned int)ga;                              // global_addr[31:0]
    g0[3] = 0x80000000u |                                  // type=2 ("image")
            (unsigned int)((ga >> 32) & 0x01FFFFFFu);      // global_addr[56:32]

    i32x8 g1;
    g1[0] = 0x00010000;                    // workgroup_mask=0, data_size=2B
    g1[1] = (int)((tdim0 & 0xFFFF) << 16); // tensor_dim0 lo16 at [31:16]
    g1[2] = (int)((tdim0 >> 16) & 0xFFFF) | (int)((tdim1 & 0xFFFF) << 16);
    g1[3] = (int)((tdim1 >> 16) & 0xFFFF) | (int)(tile0 << 16);   // tile_dim0
    g1[4] = tile1;                         // tile_dim1, tile_dim2=0
    g1[5] = stride0;                       // tensor_dim0_stride lo32
    g1[6] = 0;
    g1[7] = 0;

    const i32x4 gz4 = {0, 0, 0, 0};
    const i32x8 gz8 = {0, 0, 0, 0, 0, 0, 0, 0};
    __builtin_amdgcn_tensor_load_to_lds(g0, g1, gz4, gz4, gz8, 0);
}

// ---------------------------------------------------------------------------
// Kernel 2: fused flash attention. One wave32 per (batch, 16-query tile);
// 8 waves of a block share one batch: V (64x128 2D tile) and pre-packed K
// operands (contiguous 8KB) are staged into LDS by the TDM, double buffered,
// synchronized with TENSORcnt + workgroup barriers.
// ---------------------------------------------------------------------------
__global__ __launch_bounds__(256) void flash_attn_kernel(
    const _Float16* __restrict__ qA, const _Float16* __restrict__ kA,
    const _Float16* __restrict__ vm, const float* __restrict__ x,
    const float* __restrict__ gamma_p, float* __restrict__ out)
{
    __shared__ alignas(16) _Float16 vbuf[2][NC * VCHUNK];     // 2 x 16KB
    __shared__ alignas(16) _Float16 kbuf[2][KCH * OPTILE];    // 2 x 8KB

    const int lane = threadIdx.x & 31;
    const int wave = threadIdx.x >> 5;
    const int l16  = lane & 15;
    const int half = lane >> 4;               // 0: lanes 0-15, 1: lanes 16-31

    const int tile = blockIdx.x * 8 + wave;   // 0..2047
    const int b    = tile >> 8;
    const int it   = tile & 255;
    const int i0   = it << 4;                 // query tile base

    // Q as B-operand: pre-packed image, one 32B load, no masking.
    v16h qb = *(const v16h*)(qA + ((size_t)(b * NTIL + it) * 32 + lane) * 16);

    v8f acc[4];                               // O^T, 4 c-tiles of 16
#pragma unroll
    for (int t = 0; t < 4; ++t)
#pragma unroll
        for (int r = 0; r < 8; ++r) acc[t][r] = 0.f;

    float m = -1e30f;                         // running row max (row i = l16)
    float l = 0.f;                            // running row sum

    const _Float16* kAb = kA + (size_t)b * NTIL * OPTILE;   // batch K operands
    const _Float16* vg  = vm + (size_t)b * NC * NPIX;       // batch V tensor

    // prime the TDM pipeline: chunk 0 -> buffers [0]
    if (wave == 0) {
        tdm_load_2d(vg, (unsigned int)(uintptr_t)&vbuf[0][0],
                    VCHUNK, NC, NPIX, NC, NPIX);
        tdm_load_2d(kAb, (unsigned int)(uintptr_t)&kbuf[0][0],
                    KCH * OPTILE, 1, KCH * OPTILE, 1, KCH * OPTILE);
    }

    for (int ch = 0; ch < NCHUNK; ++ch) {
        if (wave == 0) {
            if (ch + 1 < NCHUNK) {
                const int p = (ch + 1) & 1;
                tdm_load_2d(vg + (size_t)(ch + 1) * VCHUNK,
                            (unsigned int)(uintptr_t)&vbuf[p][0],
                            VCHUNK, NC, NPIX, NC, NPIX);
                tdm_load_2d(kAb + (size_t)(ch + 1) * KCH * OPTILE,
                            (unsigned int)(uintptr_t)&kbuf[p][0],
                            KCH * OPTILE, 1, KCH * OPTILE, 1, KCH * OPTILE);
                __builtin_amdgcn_s_wait_tensorcnt(2);   // chunk ch's pair done
            } else {
                __builtin_amdgcn_s_wait_tensorcnt(0);
            }
        }
        __syncthreads();                       // chunk ch visible to all waves
        const _Float16* vl = &vbuf[ch & 1][0];
        const _Float16* kl = &kbuf[ch & 1][0];

        // 64 j columns per softmax update: 4 S^T WMMAs, 8 PV WMMAs.
        for (int jb = 0; jb < VCHUNK; jb += 64) {
            // Hoist all 4 K-operand LDS loads so the S^T WMMAs overlap the
            // ds latency (waits become dscnt<=6/4/2/0 instead of 4x dscnt==0).
            v16h ka[4];
#pragma unroll
            for (int tt = 0; tt < 4; ++tt)
                ka[tt] = *(const v16h*)(
                    kl + ((size_t)((jb >> 4) + tt) * 32 + lane) * 16);

            v8f st[4];
            {
                v8f z = {};
#pragma unroll
                for (int tt = 0; tt < 4; ++tt)
                    st[tt] = __builtin_amdgcn_wmma_f32_16x16x32_f16(
                        false, ka[tt], false, qb, (short)0, z, false, false);
            }

            // ---- online softmax over j (M-dim of S^T) ----
            float mloc = -1e30f;
#pragma unroll
            for (int tt = 0; tt < 4; ++tt)
#pragma unroll
                for (int r = 0; r < 8; ++r) mloc = fmaxf(mloc, st[tt][r]);
            mloc = fmaxf(mloc, __shfl_xor(mloc, 16, 32));
            const float mnew  = fmaxf(m, mloc);
            const float alpha = __expf(m - mnew);

            float p[4][8];
            float lloc = 0.f;
#pragma unroll
            for (int tt = 0; tt < 4; ++tt)
#pragma unroll
                for (int r = 0; r < 8; ++r) {
                    p[tt][r] = __expf(st[tt][r] - mnew);
                    lloc += p[tt][r];
                }
            l = l * alpha + lloc + __shfl_xor(lloc, 16, 32);
            m = mnew;

#pragma unroll
            for (int t = 0; t < 4; ++t)
#pragma unroll
                for (int r = 0; r < 8; ++r) acc[t][r] *= alpha;

            // ---- P^T B-operands (32(j) x 16(i)) via half-wave swaps ----
            v16h pb01, pb23;
#pragma unroll
            for (int r = 0; r < 8; ++r) {
                const float s0 = __shfl_xor(p[0][r], 16, 32);
                const float s1 = __shfl_xor(p[1][r], 16, 32);
                pb01[r]     = (_Float16)(half ? s1 : p[0][r]);
                pb01[r + 8] = (_Float16)(half ? p[1][r] : s0);
                const float s2 = __shfl_xor(p[2][r], 16, 32);
                const float s3 = __shfl_xor(p[3][r], 16, 32);
                pb23[r]     = (_Float16)(half ? s3 : p[2][r]);
                pb23[r + 8] = (_Float16)(half ? p[3][r] : s2);
            }

            // ---- O^T += V^T x P^T, V^T tiles read from LDS ----
#pragma unroll
            for (int t = 0; t < 4; ++t) {
                const _Float16* vrow =
                    vl + (size_t)(t * 16 + l16) * VCHUNK + jb + half * 16;
                v8h lo0 = *(const v8h*)(vrow);        // j = jb+half*16+0..7
                v8h hi0 = *(const v8h*)(vrow + 8);    // j = jb+half*16+8..15
                v8h lo1 = *(const v8h*)(vrow + 32);   // j = jb+32+half*16+0..7
                v8h hi1 = *(const v8h*)(vrow + 40);   // j = jb+32+half*16+8..15
                v16h va01, va23;
#pragma unroll
                for (int e = 0; e < 8; ++e) {
                    va01[e] = lo0[e]; va01[e + 8] = hi0[e];
                    va23[e] = lo1[e]; va23[e + 8] = hi1[e];
                }
                acc[t] = __builtin_amdgcn_wmma_f32_16x16x32_f16(
                    false, va01, false, pb01, (short)0, acc[t], false, false);
                acc[t] = __builtin_amdgcn_wmma_f32_16x16x32_f16(
                    false, va23, false, pb23, (short)0, acc[t], false, false);
            }
        }
        __syncthreads();   // everyone done with buffers [ch&1] before overwrite
    }

    // ---- epilogue: out = gamma * O / l + x ----
    const float g  = gamma_p[0];
    const float rl = 1.f / l;
    const int   i  = i0 + l16;
#pragma unroll
    for (int t = 0; t < 4; ++t) {
#pragma unroll
        for (int r = 0; r < 8; ++r) {
            const int    c   = t * 16 + r + half * 8;   // D layout: M=r+8*half
            const size_t off = ((size_t)b * NC + c) * NPIX + i;
            out[off] = g * acc[t][r] * rl + x[off];
        }
    }
}

// ---------------------------------------------------------------------------
extern "C" void kernel_launch(void* const* d_in, const int* in_sizes, int n_in,
                              void* d_out, int out_size, void* d_ws, size_t ws_size,
                              hipStream_t stream)
{
    const float* x     = (const float*)d_in[0];
    const float* wq    = (const float*)d_in[1];
    const float* bq    = (const float*)d_in[2];
    const float* wk    = (const float*)d_in[3];
    const float* bk    = (const float*)d_in[4];
    const float* wv    = (const float*)d_in[5];
    const float* bv    = (const float*)d_in[6];
    const float* gamma = (const float*)d_in[7];
    float*       out   = (float*)d_out;

    // workspace (f16): qA[8][256][32][16] (2MB), kA same (2MB), v[8][64][4096] (4MB)
    _Float16* qA = (_Float16*)d_ws;
    _Float16* kA = qA + (size_t)NB * NTIL * OPTILE;
    _Float16* vm = kA + (size_t)NB * NTIL * OPTILE;

    qkv_proj_kernel<<<(NB * NPIX) / 256, 256, 0, stream>>>(
        x, wq, bq, wk, bk, wv, bv, qA, kA, vm);

    // 2048 wave-tiles (8 batches x 256 query tiles), 8 waves per block
    flash_attn_kernel<<<256, 256, 0, stream>>>(qA, kA, vm, x, gamma, out);
}
